// APPNPNet_61229053772417
// MI455X (gfx1250) — matrix operand order repaired and checked by reading, not compiled, not run on hardware
//
#include <hip/hip_runtime.h>
#include <hip/hip_bf16.h>

#define D_IN  256
#define D_H   64
#define ALPHA 0.1f
#define KHOPS 10
#define APAD  8                 // LDS row padding (elements): row stride 264*2B = 528B, 16B-aligned

typedef __attribute__((ext_vector_type(16))) __bf16 v16bf;
typedef __attribute__((ext_vector_type(8)))  __bf16 v8bf;
typedef __attribute__((ext_vector_type(4)))  __bf16 v4bf;
typedef __attribute__((ext_vector_type(8)))  float  v8f;

// Split f32 into bf16 hi + bf16 lo residual using the compiler's native converts.
__device__ __forceinline__ __bf16 bf_hi(float v) { return (__bf16)v; }
__device__ __forceinline__ __bf16 bf_lo(float v, __bf16 h) { return (__bf16)(v - (float)h); }

// Split W1 [256 x 64] into transposed bf16 hi/lo planes Wt[n][k], n=0..63, k=0..255.
__global__ void split_w1_kernel(const float* __restrict__ W1,
                                __bf16* __restrict__ wt_hi, __bf16* __restrict__ wt_lo) {
  int t = blockIdx.x * blockDim.x + threadIdx.x;   // 0 .. 16383
  int k = t >> 6, n = t & 63;
  float w = W1[k * D_H + n];
  __bf16 h = bf_hi(w);
  wt_hi[n * D_IN + k] = h;
  wt_lo[n * D_IN + k] = bf_lo(w, h);
}

// h0 = relu(X @ W1 + b1), also seeds h_a = h0.
// One 128-thread block (4 waves) per 16-row M tile; wave w handles N-tile w.
// The 16x256 A panel is staged & split to bf16 hi/lo in LDS ONCE per block;
// all 4 waves consume identical A fragments from LDS.
// Split-bf16: X@W ~= Xhi@Whi + Xhi@Wlo + Xlo@Whi via v_wmma_f32_16x16x32_bf16.
__global__ void gemm1_kernel(const float* __restrict__ x,
                             const __bf16* __restrict__ wt_hi,
                             const __bf16* __restrict__ wt_lo,
                             const float* __restrict__ b1,
                             float* __restrict__ h0, float* __restrict__ h_a) {
  __shared__ __bf16 a_hi[16][D_IN + APAD];
  __shared__ __bf16 a_lo[16][D_IN + APAD];

  const int tid    = threadIdx.x;
  const int m_tile = blockIdx.x;

  // ---- Stage & split A panel (16 x 256 f32 -> two bf16 panels), coalesced float4 ----
  #pragma unroll
  for (int j = 0; j < 8; ++j) {
    int idx4 = j * 128 + tid;            // 0..1023 float4s
    int m    = idx4 >> 6;                // 64 float4 per row
    int k    = (idx4 & 63) << 2;
    const float4 v = *(const float4*)(x + (size_t)(m_tile * 16 + m) * D_IN + k);
    v4bf h4, l4;
    __bf16 hx = bf_hi(v.x);  h4[0] = hx;  l4[0] = bf_lo(v.x, hx);
    __bf16 hy = bf_hi(v.y);  h4[1] = hy;  l4[1] = bf_lo(v.y, hy);
    __bf16 hz = bf_hi(v.z);  h4[2] = hz;  l4[2] = bf_lo(v.z, hz);
    __bf16 hw = bf_hi(v.w);  h4[3] = hw;  l4[3] = bf_lo(v.w, hw);
    *(v4bf*)&a_hi[m][k] = h4;
    *(v4bf*)&a_lo[m][k] = l4;
  }
  __syncthreads();

  // ---- Per-wave 16x16 WMMA tile ----
  const int wave = tid >> 5;
  const int lane = tid & 31;
  const int n_tile = wave;
  const int ln16 = lane & 15;            // M for A, N for B/C/D
  const int hi16 = lane >> 4;
  const int h8   = hi16 << 3;

  const int     n_g     = n_tile * 16 + ln16;
  const __bf16* bh_base = wt_hi + (size_t)n_g * D_IN + hi16 * 16;
  const __bf16* bl_base = wt_lo + (size_t)n_g * D_IN + hi16 * 16;

  v8f c = {};
  #pragma unroll
  for (int kk = 0; kk < D_IN; kk += 32) {
    // A fragment: lane holds K = kk+hi16*16..+15 (elems 0..15), matching B
    v8bf ah0 = *(const v8bf*)&a_hi[ln16][kk + hi16 * 16];
    v8bf ah1 = *(const v8bf*)&a_hi[ln16][kk + hi16 * 16 + 8];
    v8bf al0 = *(const v8bf*)&a_lo[ln16][kk + hi16 * 16];
    v8bf al1 = *(const v8bf*)&a_lo[ln16][kk + hi16 * 16 + 8];
    v16bf ah = __builtin_shufflevector(ah0, ah1, 0,1,2,3,4,5,6,7,8,9,10,11,12,13,14,15);
    v16bf al = __builtin_shufflevector(al0, al1, 0,1,2,3,4,5,6,7,8,9,10,11,12,13,14,15);
    // B fragment: 16 contiguous bf16 (K = kk + hi16*16 + 2v,2v+1)
    v16bf bh = *(const v16bf*)(bh_base + kk);
    v16bf bl = *(const v16bf*)(bl_base + kk);
    c = __builtin_amdgcn_wmma_f32_16x16x32_bf16(false, ah, false, bh, (short)0, c, false, false);
    c = __builtin_amdgcn_wmma_f32_16x16x32_bf16(false, ah, false, bl, (short)0, c, false, false);
    c = __builtin_amdgcn_wmma_f32_16x16x32_bf16(false, al, false, bh, (short)0, c, false, false);
  }

  const float bias = b1[n_g];
  #pragma unroll
  for (int r = 0; r < 8; ++r) {
    int m_g  = m_tile * 16 + r + h8;     // C/D layout: VGPR r -> M=r (lanes 0-15), M=8+r (16-31)
    float v  = c[r] + bias;
    v        = fmaxf(v, 0.0f);
    size_t o = (size_t)m_g * D_H + n_g;
    h0[o]  = v;
    h_a[o] = v;
  }
}

__global__ void deg_init_kernel(float* __restrict__ deg, int N) {
  int i = blockIdx.x * blockDim.x + threadIdx.x;
  if (i < N) deg[i] = 1.0f;                   // self loop
}

__global__ void deg_acc_kernel(const int* __restrict__ col, float* __restrict__ deg, int E) {
  int e = blockIdx.x * blockDim.x + threadIdx.x;
  if (e < E) atomicAdd(deg + col[e], 1.0f);
}

__global__ void rsqrt_kernel(float* __restrict__ deg, int N) {
  int i = blockIdx.x * blockDim.x + threadIdx.x;
  if (i < N) {
    float d = deg[i];
    deg[i] = (d > 0.0f) ? rsqrtf(d) : 0.0f;   // deg now holds dinv
  }
}

__global__ void norm_kernel(const int* __restrict__ row, const int* __restrict__ col,
                            const float* __restrict__ dinv, float* __restrict__ norm, int E) {
  int e = blockIdx.x * blockDim.x + threadIdx.x;
  if (e < E) norm[e] = dinv[row[e]] * dinv[col[e]];
}

// h_next = ALPHA*h0 + (1-ALPHA)*dinv_i^2*h_cur   (teleport + self-loop term, no atomics)
__global__ void combine_init_kernel(const float* __restrict__ h0,
                                    const float* __restrict__ h_cur,
                                    const float* __restrict__ dinv,
                                    float* __restrict__ h_next, int total) {
  int t = blockIdx.x * blockDim.x + threadIdx.x;
  if (t < total) {
    float di = dinv[t >> 6];
    h_next[t] = ALPHA * h0[t] + (1.0f - ALPHA) * di * di * h_cur[t];
  }
}

// h_next[col] += (1-ALPHA)*norm[e]*h_cur[row]; 16 threads per edge, float4 feature slabs.
__global__ void scatter_kernel(const int* __restrict__ row, const int* __restrict__ col,
                               const float* __restrict__ norm,
                               const float* __restrict__ h_cur,
                               float* __restrict__ h_next, int E) {
  int g = blockIdx.x * blockDim.x + threadIdx.x;
  int e = g >> 4;
  if (e >= E) return;
  int q = (g & 15) << 2;
  float  c = (1.0f - ALPHA) * norm[e];
  int    r = row[e];
  int    w = col[e];
  const float4 v = *(const float4*)(h_cur + (size_t)r * D_H + q);
  float* dst = h_next + (size_t)w * D_H + q;
  atomicAdd(dst + 0, c * v.x);
  atomicAdd(dst + 1, c * v.y);
  atomicAdd(dst + 2, c * v.z);
  atomicAdd(dst + 3, c * v.w);
}

// out[i] = h[i,:] . W2 + b2 ; one wave32 per node, shuffle reduction.
__global__ void out_kernel(const float* __restrict__ h, const float* __restrict__ W2,
                           const float* __restrict__ b2, float* __restrict__ out, int N) {
  int wave = threadIdx.x >> 5;
  int lane = threadIdx.x & 31;
  int node = blockIdx.x * 8 + wave;
  if (node >= N) return;
  const float* hr = h + (size_t)node * D_H;
  float p = hr[lane] * W2[lane] + hr[lane + 32] * W2[lane + 32];
  #pragma unroll
  for (int off = 16; off; off >>= 1) p += __shfl_xor(p, off, 32);
  if (lane == 0) out[node] = p + b2[0];
}

extern "C" void kernel_launch(void* const* d_in, const int* in_sizes, int n_in,
                              void* d_out, int out_size, void* d_ws, size_t ws_size,
                              hipStream_t stream) {
  const float* x   = (const float*)d_in[0];
  const int*   ei  = (const int*)d_in[1];
  const float* W1  = (const float*)d_in[2];
  const float* b1  = (const float*)d_in[3];
  const float* W2  = (const float*)d_in[4];
  const float* b2  = (const float*)d_in[5];
  float*       out = (float*)d_out;

  const int N = in_sizes[0] / D_IN;     // 100000
  const int E = in_sizes[1] / 2;        // 1600000
  const int* row = ei;
  const int* col = ei + E;
  const size_t N64 = (size_t)N * D_H;

  // Workspace carve-up (all region sizes multiples of 4 floats -> 16B alignment)
  float* deg  = (float*)d_ws;                       // N  (becomes dinv)
  size_t degCap = ((size_t)N + 31) & ~(size_t)31;
  float* nrm  = deg + degCap;                       // E
  size_t eCap = ((size_t)E + 3) & ~(size_t)3;
  float* h0   = nrm + eCap;                         // N*64
  float* h_a  = h0 + N64;                           // N*64
  float* h_b  = h_a + N64;                          // N*64
  __bf16* wt_hi = (__bf16*)(h_b + N64);             // 64*256
  __bf16* wt_lo = wt_hi + (size_t)D_IN * D_H;       // 64*256

  const int B = 256;
  // 1) Pre-split W1 into transposed bf16 hi/lo planes
  split_w1_kernel<<<(D_IN * D_H) / B, B, 0, stream>>>(W1, wt_hi, wt_lo);
  // 2) h0 = relu(X@W1 + b1) via split-bf16 WMMA (LDS-staged A panel); seeds h_a
  gemm1_kernel<<<N / 16, 128, 0, stream>>>(x, wt_hi, wt_lo, b1, h0, h_a);
  // 3) GCN symmetric norm with self loops
  deg_init_kernel<<<(N + B - 1) / B, B, 0, stream>>>(deg, N);
  deg_acc_kernel<<<(E + B - 1) / B, B, 0, stream>>>(col, deg, E);
  rsqrt_kernel<<<(N + B - 1) / B, B, 0, stream>>>(deg, N);
  norm_kernel<<<(E + B - 1) / B, B, 0, stream>>>(row, col, deg, nrm, E);
  // 4) K rounds of propagation (ping-pong h_a/h_b; L2-resident)
  float* cur = h_a;
  float* nxt = h_b;
  const int total = (int)N64;
  for (int k = 0; k < KHOPS; ++k) {
    combine_init_kernel<<<(total + B - 1) / B, B, 0, stream>>>(h0, cur, deg, nxt, total);
    scatter_kernel<<<(int)(((size_t)E * 16 + B - 1) / B), B, 0, stream>>>(row, col, nrm, cur, nxt, E);
    float* t = cur; cur = nxt; nxt = t;
  }
  // 5) Final projection
  out_kernel<<<(N + 7) / 8, B, 0, stream>>>(cur, W2, b2, out, N);
}
